// ExpertChoiceMoELayer_68899865362459
// MI455X (gfx1250) — matrix operand order repaired and checked by reading, not compiled, not run on hardware
//
#include <hip/hip_runtime.h>
#include <math.h>

#define TOK 16384
#define DIM 1024
#define FF  2048
#define NE  8
#define CAP 2048

typedef __attribute__((ext_vector_type(16))) __bf16    bf16x16;
typedef __attribute__((ext_vector_type(8)))  float     f32x8;
typedef __attribute__((ext_vector_type(4)))  unsigned  u32x4;

union Frag { u32x4 q[2]; bf16x16 v; };

__device__ __forceinline__ unsigned short f32_to_bf16(float f) {
  unsigned u = __float_as_uint(f);
  u += 0x7FFFu + ((u >> 16) & 1u);   // round-to-nearest-even
  return (unsigned short)(u >> 16);
}

// ---------------- x: f32 -> bf16 (row-major kept) ----------------
__global__ void k_cvt_x(const float* __restrict__ x, unsigned short* __restrict__ xb) {
  size_t n = (size_t)TOK * DIM;
  for (size_t i = blockIdx.x * (size_t)blockDim.x + threadIdx.x; i < n;
       i += (size_t)gridDim.x * blockDim.x)
    xb[i] = f32_to_bf16(x[i]);
}

// ---------- transpose+convert: src[z][R][C] f32 -> dst[z][C][R] bf16 ----------
__global__ void k_transpose(const float* __restrict__ src, unsigned short* __restrict__ dst,
                            int R, int C) {
  __shared__ unsigned short tile[32][33];
  int z = blockIdx.z;
  const float* s = src + (size_t)z * R * C;
  unsigned short* d = dst + (size_t)z * R * C;
  int c0 = blockIdx.x * 32, r0 = blockIdx.y * 32;
  for (int i = threadIdx.y; i < 32; i += 8)
    tile[i][threadIdx.x] = f32_to_bf16(s[(size_t)(r0 + i) * C + c0 + threadIdx.x]);
  __syncthreads();
  for (int i = threadIdx.y; i < 32; i += 8)
    d[(size_t)(c0 + i) * R + r0 + threadIdx.x] = tile[threadIdx.x][i];
}

// ---------------- router: logits = x@Wg, softmax -> probs[T][8] ----------------
__global__ __launch_bounds__(128) void k_router(const float* __restrict__ x,
                                                const float* __restrict__ Wg,
                                                float* __restrict__ probs) {
  const int t = blockIdx.x * 4 + (threadIdx.x >> 5);   // one wave per token
  const int lane = threadIdx.x & 31;
  const float* xr = x + (size_t)t * DIM;
  float acc[NE];
#pragma unroll
  for (int e = 0; e < NE; e++) acc[e] = 0.0f;
  for (int d = lane; d < DIM; d += 32) {
    float xv = xr[d];
    const float* wr = Wg + d * NE;
#pragma unroll
    for (int e = 0; e < NE; e++) acc[e] += xv * wr[e];
  }
#pragma unroll
  for (int e = 0; e < NE; e++) {
#pragma unroll
    for (int off = 16; off > 0; off >>= 1) acc[e] += __shfl_xor(acc[e], off, 32);
  }
  if (lane == 0) {
    float m = acc[0];
#pragma unroll
    for (int e = 1; e < NE; e++) m = fmaxf(m, acc[e]);
    float s = 0.0f, ex[NE];
#pragma unroll
    for (int e = 0; e < NE; e++) { ex[e] = expf(acc[e] - m); s += ex[e]; }
    float inv = 1.0f / s;
#pragma unroll
    for (int e = 0; e < NE; e++) probs[(size_t)t * NE + e] = ex[e] * inv;
  }
}

// ---------------- init scratch (counts=0, lists=-1) ----------------
__global__ void k_init(int* counts, int* lists) {
  int i = blockIdx.x * blockDim.x + threadIdx.x;
  if (i < NE) counts[i] = 0;
  for (size_t j = i; j < (size_t)NE * TOK; j += (size_t)gridDim.x * blockDim.x)
    lists[j] = -1;
}

// -------- per-expert radix select of rank-CAP prob (threshold τ_e) --------
__global__ void k_select(const float* __restrict__ probs, unsigned* __restrict__ thr) {
  const int e = blockIdx.x;
  __shared__ int hist[256];
  __shared__ unsigned s_prefix;
  __shared__ int s_krem;
  if (threadIdx.x == 0) { s_prefix = 0u; s_krem = CAP; }
  for (int byte = 3; byte >= 0; --byte) {
    __syncthreads();
    for (int i = threadIdx.x; i < 256; i += blockDim.x) hist[i] = 0;
    __syncthreads();
    const unsigned prefix = s_prefix;
    for (int t = threadIdx.x; t < TOK; t += blockDim.x) {
      unsigned u = __float_as_uint(probs[(size_t)t * NE + e]);   // positive floats: bit order == value order
      bool match = (byte == 3) ? true
                               : ((u >> (8 * (byte + 1))) == (prefix >> (8 * (byte + 1))));
      if (match) atomicAdd(&hist[(u >> (8 * byte)) & 255], 1);
    }
    __syncthreads();
    if (threadIdx.x == 0) {
      int krem = s_krem;
      int b = 255;
      while (b > 0) {
        int c = hist[b];
        if (c >= krem) break;
        krem -= c; --b;
      }
      s_prefix = prefix | ((unsigned)b << (8 * byte));
      s_krem = krem;
    }
  }
  __syncthreads();
  if (threadIdx.x == 0) thr[e] = s_prefix;
}

// ------ per-token expert assignment + weight + per-expert token lists ------
__global__ void k_assign(const float* __restrict__ probs, const unsigned* __restrict__ thr,
                         int* counts, int* lists, float* __restrict__ wts) {
  int t = blockIdx.x * blockDim.x + threadIdx.x;
  if (t >= TOK) return;
  float p[NE];
#pragma unroll
  for (int e = 0; e < NE; e++) p[e] = probs[(size_t)t * NE + e];
  int best = -1; float bw = -1.0f;
#pragma unroll
  for (int e = 0; e < NE; e++) {
    float tau = __uint_as_float(thr[e]);
    if (p[e] >= tau && p[e] > bw) { bw = p[e]; best = e; }   // strict > => lowest expert wins ties
  }
  if (best < 0) {
#pragma unroll
    for (int e = 0; e < NE; e++) if (p[e] > bw) { bw = p[e]; best = e; }
  }
  wts[t] = bw;
  int slot = atomicAdd(&counts[best], 1);
  lists[best * TOK + slot] = t;
}

// ---------------- padded (to 32 rows) prefix offsets into compact H ----------------
__global__ void k_offsets(const int* counts, int* offs) {
  if (threadIdx.x == 0 && blockIdx.x == 0) {
    int o = 0;
    for (int e = 0; e < NE; e++) { offs[e] = o; o += (counts[e] + 31) & ~31; }
    offs[NE] = o;
  }
}

__device__ __forceinline__ Frag load_a(const unsigned short* base, int k0, int khalf) {
  Frag a;
  const unsigned short* p = base + k0 + khalf * 16;
  a.q[0] = *(const u32x4*)p;          // K [khalf*16, +8)
  a.q[1] = *(const u32x4*)(p + 8);    // K [khalf*16+8, +8)
  return a;
}

// ------------- GEMM1: H = gelu(gather(x) @ W1[e]), bf16 WMMA, 32x64/wave -------------
__global__ __launch_bounds__(128) void k_ffn1(const unsigned short* __restrict__ xb,
                                              const unsigned short* __restrict__ w1t,
                                              const int* __restrict__ lists,
                                              const int* __restrict__ counts,
                                              const int* __restrict__ offs,
                                              unsigned short* __restrict__ H) {
  const int e = blockIdx.z;
  const int cnt = counts[e];
  const int padded = (cnt + 31) & ~31;
  const int mbase = blockIdx.y * 32;
  if (mbase >= padded) return;
  const int wave = threadIdx.x >> 5, lane = threadIdx.x & 31;
  const int row = lane & 15, khalf = lane >> 4;
  const int tid0 = lists[e * TOK + mbase + row];
  const int tid1 = lists[e * TOK + mbase + 16 + row];
  const unsigned short* xr0 = (tid0 >= 0) ? xb + (size_t)tid0 * DIM : nullptr;
  const unsigned short* xr1 = (tid1 >= 0) ? xb + (size_t)tid1 * DIM : nullptr;
  const int nb = blockIdx.x * 256 + wave * 64;
  const int col = nb + row;
  const unsigned short* wb = w1t + (size_t)e * FF * DIM;   // [F][D] (pre-transposed)
  f32x8 acc[2][4];
#pragma unroll
  for (int mt = 0; mt < 2; mt++)
#pragma unroll
    for (int nt = 0; nt < 4; nt++) acc[mt][nt] = (f32x8)0.0f;

  for (int k0 = 0; k0 < DIM; k0 += 32) {
    Frag a0, a1;
    if (xr0) a0 = load_a(xr0, k0, khalf); else { a0.q[0] = (u32x4)0u; a0.q[1] = (u32x4)0u; }
    if (xr1) a1 = load_a(xr1, k0, khalf); else { a1.q[0] = (u32x4)0u; a1.q[1] = (u32x4)0u; }
    if (k0 + 32 < DIM) {                       // gfx1250 global_prefetch_b8, one K-step ahead
      if (xr0) __builtin_prefetch(xr0 + k0 + 32, 0, 1);
      if (xr1) __builtin_prefetch(xr1 + k0 + 32, 0, 1);
    }
#pragma unroll
    for (int nt = 0; nt < 4; nt++) {
      const unsigned short* bp = wb + (size_t)(col + nt * 16) * DIM + k0 + khalf * 16;
      Frag b;
      b.q[0] = *(const u32x4*)bp;              // K [khalf*16, +8)
      b.q[1] = *(const u32x4*)(bp + 8);        // K [khalf*16+8, +8)
      if (k0 + 32 < DIM) __builtin_prefetch(bp + 32, 0, 1);
      acc[0][nt] = __builtin_amdgcn_wmma_f32_16x16x32_bf16(
          false, a0.v, false, b.v, (short)0, acc[0][nt], false, false);
      acc[1][nt] = __builtin_amdgcn_wmma_f32_16x16x32_bf16(
          false, a1.v, false, b.v, (short)0, acc[1][nt], false, false);
    }
  }
#pragma unroll
  for (int mt = 0; mt < 2; mt++) {
    const int hrow = offs[e] + mbase + mt * 16;
#pragma unroll
    for (int nt = 0; nt < 4; nt++) {
#pragma unroll
      for (int v = 0; v < 8; v++) {
        int M = v + 8 * khalf;
        float g = acc[mt][nt][v];
        g = 0.5f * g * (1.0f + erff(g * 0.70710678118f));   // exact gelu
        H[(size_t)(hrow + M) * FF + (nb + nt * 16 + row)] = f32_to_bf16(g);
      }
    }
  }
}

// ---- GEMM2: out[tid] = wts[tid] * (H @ W2[e]) scatter, bf16 WMMA, 32x64/wave ----
__global__ __launch_bounds__(128) void k_ffn2(const unsigned short* __restrict__ H,
                                              const unsigned short* __restrict__ w2t,
                                              const int* __restrict__ lists,
                                              const int* __restrict__ counts,
                                              const int* __restrict__ offs,
                                              const float* __restrict__ wts,
                                              float* __restrict__ out) {
  const int e = blockIdx.z;
  const int cnt = counts[e];
  const int padded = (cnt + 31) & ~31;
  const int mbase = blockIdx.y * 32;
  if (mbase >= padded) return;
  const int wave = threadIdx.x >> 5, lane = threadIdx.x & 31;
  const int row = lane & 15, khalf = lane >> 4;
  const int nb = blockIdx.x * 256 + wave * 64;
  const int col = nb + row;
  const int hrow = offs[e] + mbase;
  const unsigned short* ar0 = H + (size_t)(hrow + row) * FF;
  const unsigned short* ar1 = H + (size_t)(hrow + 16 + row) * FF;
  const unsigned short* wb = w2t + (size_t)e * DIM * FF;   // [D][F] (pre-transposed)
  f32x8 acc[2][4];
#pragma unroll
  for (int mt = 0; mt < 2; mt++)
#pragma unroll
    for (int nt = 0; nt < 4; nt++) acc[mt][nt] = (f32x8)0.0f;

  for (int k0 = 0; k0 < FF; k0 += 32) {
    Frag a0 = load_a(ar0, k0, khalf);
    Frag a1 = load_a(ar1, k0, khalf);
    if (k0 + 32 < FF) {
      __builtin_prefetch(ar0 + k0 + 32, 0, 1);
      __builtin_prefetch(ar1 + k0 + 32, 0, 1);
    }
#pragma unroll
    for (int nt = 0; nt < 4; nt++) {
      const unsigned short* bp = wb + (size_t)(col + nt * 16) * FF + k0 + khalf * 16;
      Frag b;
      b.q[0] = *(const u32x4*)bp;
      b.q[1] = *(const u32x4*)(bp + 8);
      if (k0 + 32 < FF) __builtin_prefetch(bp + 32, 0, 1);
      acc[0][nt] = __builtin_amdgcn_wmma_f32_16x16x32_bf16(
          false, a0.v, false, b.v, (short)0, acc[0][nt], false, false);
      acc[1][nt] = __builtin_amdgcn_wmma_f32_16x16x32_bf16(
          false, a1.v, false, b.v, (short)0, acc[1][nt], false, false);
    }
  }
  int tids[2][8]; float ww[2][8];
#pragma unroll
  for (int mt = 0; mt < 2; mt++) {
#pragma unroll
    for (int v = 0; v < 8; v++) {
      int M = v + 8 * khalf;
      int t2 = lists[e * TOK + mbase + mt * 16 + M];
      tids[mt][v] = t2;
      ww[mt][v] = (t2 >= 0) ? wts[t2] : 0.0f;
    }
  }
#pragma unroll
  for (int mt = 0; mt < 2; mt++) {
#pragma unroll
    for (int nt = 0; nt < 4; nt++) {
#pragma unroll
      for (int v = 0; v < 8; v++) {
        if (tids[mt][v] >= 0)
          out[(size_t)tids[mt][v] * DIM + (nb + nt * 16 + row)] = ww[mt][v] * acc[mt][nt][v];
      }
    }
  }
}

extern "C" void kernel_launch(void* const* d_in, const int* in_sizes, int n_in,
                              void* d_out, int out_size, void* d_ws, size_t ws_size,
                              hipStream_t stream) {
  (void)in_sizes; (void)n_in; (void)out_size; (void)ws_size;
  const float* x  = (const float*)d_in[0];
  const float* Wg = (const float*)d_in[1];
  const float* W1 = (const float*)d_in[2];
  const float* W2 = (const float*)d_in[3];
  float* out = (float*)d_out;

  char* ws = (char*)d_ws;
  size_t o = 0;
  auto carve = [&](size_t bytes) -> char* {
    char* p = ws + o;
    o += (bytes + 255) & ~(size_t)255;
    return p;
  };
  float*          probs = (float*)carve((size_t)TOK * NE * 4);
  unsigned*       thr   = (unsigned*)carve(NE * 4);
  int*            counts= (int*)carve(NE * 4);
  int*            offs  = (int*)carve((NE + 1) * 4);
  int*            lists = (int*)carve((size_t)NE * TOK * 4);
  float*          wts   = (float*)carve((size_t)TOK * 4);
  unsigned short* xb    = (unsigned short*)carve((size_t)TOK * DIM * 2);
  unsigned short* w1t   = (unsigned short*)carve((size_t)NE * DIM * FF * 2);
  unsigned short* w2t   = (unsigned short*)carve((size_t)NE * DIM * FF * 2);
  unsigned short* H     = (unsigned short*)carve((size_t)(TOK + 256) * FF * 2);

  k_cvt_x<<<4096, 256, 0, stream>>>(x, xb);
  k_transpose<<<dim3(FF / 32, DIM / 32, NE), dim3(32, 8), 0, stream>>>(W1, w1t, DIM, FF);
  k_transpose<<<dim3(DIM / 32, FF / 32, NE), dim3(32, 8), 0, stream>>>(W2, w2t, FF, DIM);
  k_router<<<TOK / 4, 128, 0, stream>>>(x, Wg, probs);
  k_init<<<256, 256, 0, stream>>>(counts, lists);
  k_select<<<NE, 256, 0, stream>>>(probs, thr);
  k_assign<<<TOK / 256, 256, 0, stream>>>(probs, thr, counts, lists, wts);
  k_offsets<<<1, 1, 0, stream>>>(counts, offs);
  k_ffn1<<<dim3(FF / 256, TOK / 32, NE), 128, 0, stream>>>(xb, w1t, lists, counts, offs, H);
  k_ffn2<<<dim3(DIM / 256, TOK / 32, NE), 128, 0, stream>>>(H, w2t, lists, counts, offs, wts, out);
}
